// SphericalConv_36189394436282
// MI455X (gfx1250) — matrix operand order repaired and checked by reading, not compiled, not run on hardware
//
#include <hip/hip_runtime.h>
#include <math.h>

#define L1C  16
#define BB   4
#define NN   8192
#define CINC 64
#define COUTC 64
#define NCH  128
#define SLOPEC 0.01f

typedef _Float16 f16;
typedef __attribute__((ext_vector_type(16))) _Float16 v16h;
typedef __attribute__((ext_vector_type(8)))  float    v8f;

// ---------------- WMMA fragment swizzles (ISA 7.12.2, wave32) ----------------
// Fragment buffer = 512 f16, lane-major: element (lane, e) at [lane*16 + e].
// A 16x32 f16: lane = m + 16*((k>>3)&1);  e = (k&7) + 8*(k>>4)
__device__ __forceinline__ int a_pos(int m, int k) {
  return (m + (((k >> 3) & 1) << 4)) * 16 + (k & 7) + ((k >> 4) << 3);
}
// B 32x16 f16: lane = n + 16*((k>>3)&1);  e = (k&7) + 8*(k>>4)
__device__ __forceinline__ int b_pos(int k, int n) {
  return (n + (((k >> 3) & 1) << 4)) * 16 + (k & 7) + ((k >> 4) << 3);
}
// Fragment load: two 16B LDS loads (ds_load_b128 x2)
__device__ __forceinline__ v16h frag_ld(const f16* buf, int lane) {
  const uint4* p = (const uint4*)(buf + lane * 16);
  union { uint4 u[2]; v16h v; } r;
  r.u[0] = p[0];
  r.u[1] = p[1];
  return r.v;
}

__device__ __forceinline__ float lrelu(float x) { return x > 0.f ? x : SLOPEC * x; }

// ---------------- Legendre / angle recursion (per point, registers) ----------------
struct BasisState {
  float x, s, pmm, invf;
  float cm[L1C], sm[L1C];
  float Pm1[L1C], Pm2[L1C], rat[L1C];
};

__device__ __forceinline__ void basis_init(BasisState& st, float theta, float phi) {
  st.x = cosf(theta);
  st.s = sqrtf(fmaxf(0.f, 1.f - st.x * st.x));
  st.pmm = 1.f;
  st.invf = 1.f;
  float c1 = cosf(phi), s1 = sinf(phi);
  st.cm[0] = 1.f; st.sm[0] = 0.f;
#pragma unroll
  for (int m = 1; m < L1C; ++m) {
    float cp = st.cm[m - 1], sp = st.sm[m - 1];
    st.cm[m] = cp * c1 - sp * s1;
    st.sm[m] = sp * c1 + cp * s1;
  }
}

// Advance to degree l; qc[m] = sqrt(c_lm)*P_lm*cos(m phi), qs[m] = ...*sin(m phi).
// Entries m>l are 0. Normalized values are O(1): f16-safe.
__device__ __forceinline__ void basis_step(BasisState& st, int l,
                                           float* qc, float* qs) {
  if (l > 0) {
    st.pmm  *= -(2.f * l - 1.f) * st.s;               // P(l,l) = -(2l-1) s P(l-1,l-1)
    st.invf *= 1.f / ((2.f * l - 1.f) * (2.f * l));   // 1/(2l)!
  }
#pragma unroll
  for (int m = 0; m < L1C; ++m) {
    float Q = 0.f;
    if (m <= l) {
      float P;
      if (m == l) {
        P = st.pmm;
        st.Pm1[m] = P; st.Pm2[m] = 0.f;
        st.rat[m] = st.invf;                          // (l-m)!/(l+m)! at l==m
      } else {
        st.rat[m] *= (float)(l - m) / (float)(l + m);
        P = ((2.f * l - 1.f) * st.x * st.Pm1[m] - (float)(l - 1 + m) * st.Pm2[m])
            / (float)(l - m);
        st.Pm2[m] = st.Pm1[m]; st.Pm1[m] = P;
      }
      float q2 = 6.283185307f * sqrtf(12.56637061f / (2.f * l + 1.f))
               * (2.f * l + 1.f) * st.rat[m] * 0.07957747155f;
      Q = sqrtf(fmaxf(q2, 0.f)) * P;
    }
    qc[m] = Q * st.cm[m];
    qs[m] = Q * st.sm[m];
  }
}

// ---------------- Kernel Z: zero the f32 split-K accumulator ----------------
__global__ void k_zero(float* p, int n) {
  int i = blockIdx.x * blockDim.x + threadIdx.x;
  if (i < n) p[i] = 0.f;
}

// ---------------- Kernel A: projection + analysis (split-K via f32 atomics) --------
// SAcc layout: [b][l][r(2)][m(16)][d(64)] f32
__global__ void __launch_bounds__(256) k_analysis(
    const float* __restrict__ theta, const float* __restrict__ phi,
    const float* __restrict__ areas, const float* __restrict__ values,
    const float* __restrict__ W, float* __restrict__ SAcc) {
  extern __shared__ char smem[];
  f16* valsA = (f16*)smem;                  // [8 mt][2 ks][512]   A-swizzled values*areas
  f16* wtB   = valsA + 8 * 2 * 512;         // [2 ks][4 dt][512]   B-swizzled W[l]
  f16* prodB = wtB + 2 * 4 * 512;           // [4 ks][4 dt][512]   B-swizzled prod
  f16* qcA   = prodB + 4 * 4 * 512;         // [4 ks][512]         A-swizzled Qc
  f16* qsA   = qcA + 4 * 512;               // [4 ks][512]         A-swizzled Qs

  const int t = threadIdx.x, lane = t & 31, wv = t >> 5;
  const int b = blockIdx.y;
  const int n0 = blockIdx.x * NCH;

  // stage values*areas, A-swizzled (areas folded once -> prod inherits it)
  for (int i = t; i < NCH * CINC; i += 256) {
    int n = i >> 6, c = i & 63;
    size_t gp = (size_t)b * NN + n0 + n;
    f16 v = (f16)(values[gp * CINC + c] * areas[gp]);
    valsA[((n >> 4) * 2 + (c >> 5)) * 512 + a_pos(n & 15, c & 31)] = v;
  }
  BasisState st;
  if (t < NCH) basis_init(st, theta[(size_t)b * NN + n0 + t], phi[(size_t)b * NN + n0 + t]);

  for (int l = 0; l < L1C; ++l) {
    __syncthreads();  // previous iteration's fragment reads complete

    // stage W[l], B-swizzled (coalesced global reads, scattered u16 LDS stores)
    for (int i = t; i < CINC * COUTC; i += 256) {
      int c = i >> 6, d = i & 63;
      f16 v = (f16)W[(size_t)l * CINC * COUTC + i];
      wtB[((c >> 5) * 4 + (d >> 4)) * 512 + b_pos(c & 31, d & 15)] = v;
    }
    if (l + 1 < L1C)
      __builtin_prefetch(&W[(size_t)(l + 1) * CINC * COUTC + t], 0, 1);  // global_prefetch_b8

    // basis row for degree l, A-swizzled (M=m, K=point)
    if (t < NCH) {
      float qcr[L1C], qsr[L1C];
      basis_step(st, l, qcr, qsr);
      const int ks = t >> 5, kl = t & 31;
#pragma unroll
      for (int m = 0; m < L1C; ++m) {
        int p = ks * 512 + a_pos(m, kl);
        qcA[p] = (f16)qcr[m];
        qsA[p] = (f16)qsr[m];
      }
    }
    __syncthreads();

    // prod = vals @ W[l] : M=128 (one 16-row tile per wave), N=64, K=64
    {
      const int mt = wv;
      const int hi = lane >> 4, nc = lane & 15;
#pragma unroll
      for (int nt = 0; nt < 4; ++nt) {
        v8f acc = {};
#pragma unroll
        for (int ks = 0; ks < 2; ++ks) {
          v16h a  = frag_ld(valsA + (mt * 2 + ks) * 512, lane);
          v16h bf = frag_ld(wtB + (ks * 4 + nt) * 512, lane);
          acc = __builtin_amdgcn_wmma_f32_16x16x32_f16(false, a, false, bf,
                                                       (short)0, acc, false, false);
        }
        // D-frag -> B-swizzled prod: lane's 8 values are one contiguous 16B run
        union { f16 h[8]; uint4 u; } pk;
#pragma unroll
        for (int v = 0; v < 8; ++v) pk.h[v] = (f16)acc[v];
        f16* dst = prodB + ((mt >> 1) * 4 + nt) * 512
                 + (nc + (hi << 4)) * 16 + (mt & 1) * 8;
        *(uint4*)dst = pk.u;
      }
    }
    __syncthreads();

    // analysis: S[r][m][d] += Q{c,s}(m,n) * prod(n,d); M=16, N=64, K=128
    {
      const int r = wv >> 2, dt = wv & 3;                  // 8 waves = 2 x 4 jobs
      const f16* A = r ? qsA : qcA;
      v8f acc = {};
#pragma unroll
      for (int ks = 0; ks < 4; ++ks) {
        v16h a  = frag_ld(A + ks * 512, lane);
        v16h bf = frag_ld(prodB + (ks * 4 + dt) * 512, lane);
        acc = __builtin_amdgcn_wmma_f32_16x16x32_f16(false, a, false, bf,
                                                     (short)0, acc, false, false);
      }
      const int hi = lane >> 4, nc = lane & 15;
      float* dst = SAcc + (((size_t)b * L1C + l) * 2 + r) * (L1C * COUTC);
#pragma unroll
      for (int v = 0; v < 8; ++v) {
        int m = v + hi * 8;
        atomicAdd(&dst[m * COUTC + dt * 16 + nc], acc[v]);
      }
    }
  }
}

// ---------------- Kernel B: fac*leaky_relu folds, emit G pre-swizzled ----------------
// G layout: [b][l][dt(4)][512] f16, B-swizzled fragments with K = r*16+m, N = d&15.
// k<16: fac*leaky(S_re[m]);  k>=16: -fac*leaky(-S_im[m])
__global__ void k_act(const float* __restrict__ SAcc, f16* __restrict__ G) {
  int idx = blockIdx.x * blockDim.x + threadIdx.x;  // 131072
  int d = idx & 63;
  int m = (idx >> 6) & 15;
  int r = (idx >> 10) & 1;
  int l = (idx >> 11) & 15;
  int b = idx >> 15;
  float S = SAcc[idx];
  float fac = (m == 0) ? 1.f : 2.f;
  float g = (r == 0) ? fac * lrelu(S) : -fac * lrelu(-S);
  int k = r * 16 + m;
  G[(((size_t)b * L1C + l) * 4 + (d >> 4)) * 512 + b_pos(k, d & 15)] = (f16)g;
}

// ---------------- Kernel C: synthesis. out(n,d) = sum_l [Qc|Qs](n,k32) @ G_l(k32,d) -----
__global__ void __launch_bounds__(256) k_synth(
    const float* __restrict__ theta, const float* __restrict__ phi,
    const f16* __restrict__ G, float* __restrict__ out) {
  extern __shared__ char smem[];
  f16* qcsA = (f16*)smem;             // [8 mt][512]  A-swizzled [Qc | Qs], K=32
  f16* gB   = qcsA + 8 * 512;         // [4 dt][512]  B-swizzled G[b][l]

  const int t = threadIdx.x, lane = t & 31, wv = t >> 5;
  const int b = blockIdx.y;
  const int n0 = blockIdx.x * NCH;

  BasisState st;
  if (t < NCH) basis_init(st, theta[(size_t)b * NN + n0 + t], phi[(size_t)b * NN + n0 + t]);

  v8f zero = {};
  v8f acc[4];
#pragma unroll
  for (int i = 0; i < 4; ++i) acc[i] = zero;

  for (int l = 0; l < L1C; ++l) {
    __syncthreads();
    // stage G[b][l]: already swizzled -> contiguous 4KB uint4 copy
    {
      const uint4* src = (const uint4*)(G + ((size_t)b * L1C + l) * 4 * 512);
      ((uint4*)gB)[t] = src[t];
      if (l + 1 < L1C)
        __builtin_prefetch(G + ((size_t)b * L1C + l + 1) * 4 * 512 + t * 8, 0, 1);
    }
    // basis for degree l: thread p's 32 values = four contiguous 16B runs in A-swizzle
    if (t < NCH) {
      float qcr[L1C], qsr[L1C];
      basis_step(st, l, qcr, qsr);
      const int mt = t >> 4, pr = t & 15;
      f16* base = qcsA + mt * 512;
      union { f16 h[8]; uint4 u; } u0, u1, u2, u3;
#pragma unroll
      for (int m = 0; m < 8; ++m) {
        u0.h[m] = (f16)qcr[m];       // k=m,      m<8 : lane=pr,    e=m
        u1.h[m] = (f16)qcr[m + 8];   // k=m+8        : lane=pr+16, e=m
        u2.h[m] = (f16)qsr[m];       // k=16+m, m<8  : lane=pr,    e=8+m
        u3.h[m] = (f16)qsr[m + 8];   // k=24+m       : lane=pr+16, e=8+m
      }
      *(uint4*)(base + pr * 16)            = u0.u;
      *(uint4*)(base + (pr + 16) * 16)     = u1.u;
      *(uint4*)(base + pr * 16 + 8)        = u2.u;
      *(uint4*)(base + (pr + 16) * 16 + 8) = u3.u;
    }
    __syncthreads();

    const int mt = wv;  // 8 n-tiles, one per wave; K=32 packs re+im
#pragma unroll
    for (int dt = 0; dt < 4; ++dt) {
      v16h a  = frag_ld(qcsA + mt * 512, lane);
      v16h bf = frag_ld(gB + dt * 512, lane);
      acc[dt] = __builtin_amdgcn_wmma_f32_16x16x32_f16(false, a, false, bf,
                                                       (short)0, acc[dt], false, false);
    }
  }

  const int hi = lane >> 4, nc = lane & 15;
#pragma unroll
  for (int dt = 0; dt < 4; ++dt)
#pragma unroll
    for (int v = 0; v < 8; ++v) {
      int n = n0 + wv * 16 + v + hi * 8;
      out[((size_t)b * NN + n) * COUTC + dt * 16 + nc] = acc[dt][v];
    }
}

extern "C" void kernel_launch(void* const* d_in, const int* in_sizes, int n_in,
                              void* d_out, int out_size, void* d_ws, size_t ws_size,
                              hipStream_t stream) {
  (void)in_sizes; (void)n_in; (void)out_size; (void)ws_size;
  const float* theta  = (const float*)d_in[0];
  const float* phi    = (const float*)d_in[1];
  const float* areas  = (const float*)d_in[2];
  const float* values = (const float*)d_in[3];
  const float* W      = (const float*)d_in[4];
  float* out = (float*)d_out;

  const int accN = BB * L1C * 2 * L1C * COUTC;      // 131072 f32
  float* SAcc = (float*)d_ws;
  f16*   G    = (f16*)(SAcc + accN);                // 131072 f16 after accumulator

  k_zero<<<accN / 256, 256, 0, stream>>>(SAcc, accN);

  size_t smemA = (size_t)(8 * 2 + 2 * 4 + 4 * 4 + 4 + 4) * 512 * 2;  // 49152 B
  k_analysis<<<dim3(NN / NCH, BB), 256, smemA, stream>>>(theta, phi, areas, values, W, SAcc);

  k_act<<<accN / 256, 256, 0, stream>>>(SAcc, G);

  size_t smemC = (size_t)(8 + 4) * 512 * 2;                          // 12288 B
  k_synth<<<dim3(NN / NCH, BB), 256, smemC, stream>>>(theta, phi, G, out);
}
